// MultiHeadAttention_21483426414789
// MI455X (gfx1250) — matrix operand order repaired and checked by reading, not compiled, not run on hardware
//
#include <hip/hip_runtime.h>

// Problem dims
#define BB   2
#define LL   1024
#define EE   1024
#define HH   16
#define DHH  64
#define RM   64          // max relative position
#define RK   127         // 2*RM - 1 rows in rel tables
#define RKP  128         // padded K for rel GEMMs

typedef __bf16 bf16_t;
typedef __attribute__((ext_vector_type(16))) __bf16 bfx16;
typedef __attribute__((ext_vector_type(8)))  __bf16 bfx8;
typedef __attribute__((ext_vector_type(8)))  float  fx8;

__device__ __forceinline__ bf16_t f2bf(float f) {
  unsigned u = __builtin_bit_cast(unsigned, f);
  unsigned r = u + 0x7FFFu + ((u >> 16) & 1u);   // round-to-nearest-even
  unsigned short h = (unsigned short)(r >> 16);
  return __builtin_bit_cast(bf16_t, h);
}

__device__ __forceinline__ fx8 wmma32(bfx16 a, bfx16 b, fx8 c) {
  return __builtin_amdgcn_wmma_f32_16x16x32_bf16(false, a, false, b, (short)0, c,
                                                 false, false);
}

// ---- fragment loaders (wave32; contiguous k order used by this file) ------------
// A 16x32 bf16: lane (hi=lane>>4, M=lane&15); elem j: K = j + 16*hi
// B 32x16 bf16: lane (hi, N=lane&15);        elem j: K = j + 16*hi

__device__ __forceinline__ bfx16 load_afrag_f32(const float* p) {
  const float4 a0 = *reinterpret_cast<const float4*>(p);
  const float4 a1 = *reinterpret_cast<const float4*>(p + 4);
  const float4 a2 = *reinterpret_cast<const float4*>(p + 8);
  const float4 a3 = *reinterpret_cast<const float4*>(p + 12);
  bfx16 a;
  a[0]=f2bf(a0.x);  a[1]=f2bf(a0.y);  a[2]=f2bf(a0.z);  a[3]=f2bf(a0.w);
  a[4]=f2bf(a1.x);  a[5]=f2bf(a1.y);  a[6]=f2bf(a1.z);  a[7]=f2bf(a1.w);
  a[8]=f2bf(a2.x);  a[9]=f2bf(a2.y);  a[10]=f2bf(a2.z); a[11]=f2bf(a2.w);
  a[12]=f2bf(a3.x); a[13]=f2bf(a3.y); a[14]=f2bf(a3.z); a[15]=f2bf(a3.w);
  return a;
}

__device__ __forceinline__ bfx16 load_afrag_bf16(const bf16_t* p) {
  bfx8 x0 = *reinterpret_cast<const bfx8*>(p);
  bfx8 x1 = *reinterpret_cast<const bfx8*>(p + 8);
  bfx16 a;
#pragma unroll
  for (int j = 0; j < 8; ++j) { a[j] = x0[j]; a[j + 8] = x1[j]; }
  return a;
}

__device__ __forceinline__ bfx16 load_bfrag_bf16(const bf16_t* p) {
  bfx8 x0 = *reinterpret_cast<const bfx8*>(p);
  bfx8 x1 = *reinterpret_cast<const bfx8*>(p + 8);
  bfx16 b;
#pragma unroll
  for (int j = 0; j < 8; ++j) { b[j] = x0[j]; b[j + 8] = x1[j]; }
  return b;
}

__device__ __forceinline__ bfx16 load_bfrag_f32(const float* p) {
  const float4 a0 = *reinterpret_cast<const float4*>(p);
  const float4 a1 = *reinterpret_cast<const float4*>(p + 4);
  const float4 a2 = *reinterpret_cast<const float4*>(p + 8);
  const float4 a3 = *reinterpret_cast<const float4*>(p + 12);
  bfx16 b;
  b[0]=f2bf(a0.x);  b[1]=f2bf(a0.y);  b[2]=f2bf(a0.z);  b[3]=f2bf(a0.w);
  b[4]=f2bf(a1.x);  b[5]=f2bf(a1.y);  b[6]=f2bf(a1.z);  b[7]=f2bf(a1.w);
  b[8]=f2bf(a2.x);  b[9]=f2bf(a2.y);  b[10]=f2bf(a2.z); b[11]=f2bf(a2.w);
  b[12]=f2bf(a3.x); b[13]=f2bf(a3.y); b[14]=f2bf(a3.z); b[15]=f2bf(a3.w);
  return b;
}

// ---- kernel: pack an ExE fp32 weight matrix into B-fragment order (bf16) ----
// layout: Wp[((ks*(E/16) + ct)*32 + lane)*16 + j], ks = K-step(32), ct = col tile(16)
__global__ __launch_bounds__(256) void k_pack_w(const float* __restrict__ W,
                                                bf16_t* __restrict__ Wp) {
  int idx  = blockIdx.x * 256 + threadIdx.x;   // (E/32)*(E/16)*32 = 65536 slots
  int lane = idx & 31;
  int rest = idx >> 5;
  int ct   = rest & 63;        // E/16
  int ks   = rest >> 6;        // E/32
  int hi = lane >> 4, ln = lane & 15;
  bf16_t* dst = Wp + (size_t)idx * 16;
#pragma unroll
  for (int j = 0; j < 16; ++j) {
    int K = ks * 32 + j + 16 * hi;
    int N = ct * 16 + ln;
    dst[j] = f2bf(W[(size_t)K * EE + N]);
  }
}

// ---- dense GEMM core: acc[4] = A(f32) x Wpack(bf16) over K=E ----------------
__device__ __forceinline__ void gemm_core(const float* __restrict__ A,
                                          const bf16_t* __restrict__ Wp,
                                          int row0, int col0, int lane,
                                          fx8 acc[4]) {
  const int hi = lane >> 4, ln = lane & 15;
  for (int k0 = 0; k0 < EE; k0 += 32) {
    const float* pa = A + (size_t)(row0 + ln) * EE + k0 + 16 * hi;
    if (k0 + 256 < EE) __builtin_prefetch(pa + 256, 0, 1);
    bfx16 af = load_afrag_f32(pa);
#pragma unroll
    for (int t = 0; t < 4; ++t) {
      const bf16_t* pb =
          Wp + ((size_t)((k0 >> 5) * (EE >> 4) + (col0 >> 4) + t) * 32 + lane) * 16;
      acc[t] = wmma32(af, load_bfrag_bf16(pb), acc[t]);
    }
  }
}

// ---- kernel: projection GEMM, bf16 output -----------------------------------
__global__ __launch_bounds__(128) void k_gemm_bf16(const float* __restrict__ A,
                                                   const bf16_t* __restrict__ Wp,
                                                   const float* __restrict__ bias,
                                                   bf16_t* __restrict__ out) {
  const int lane = threadIdx.x & 31;
  const int wave = threadIdx.x >> 5;
  const int hi = lane >> 4, ln = lane & 15;
  const int row0 = blockIdx.y * 64 + wave * 16;
  const int col0 = blockIdx.x * 64;
  fx8 acc[4] = {};
  gemm_core(A, Wp, row0, col0, lane, acc);
#pragma unroll
  for (int t = 0; t < 4; ++t) {
    const int col = col0 + t * 16 + ln;
    const float bv = bias[col];
#pragma unroll
    for (int r = 0; r < 8; ++r)
      out[(size_t)(row0 + r + 8 * hi) * EE + col] = f2bf(acc[t][r] + bv);
  }
}

// ---- kernel: final GEMM, f32 output -----------------------------------------
__global__ __launch_bounds__(128) void k_gemm_f32(const float* __restrict__ A,
                                                  const bf16_t* __restrict__ Wp,
                                                  const float* __restrict__ bias,
                                                  float* __restrict__ out) {
  const int lane = threadIdx.x & 31;
  const int wave = threadIdx.x >> 5;
  const int hi = lane >> 4, ln = lane & 15;
  const int row0 = blockIdx.y * 64 + wave * 16;
  const int col0 = blockIdx.x * 64;
  fx8 acc[4] = {};
  gemm_core(A, Wp, row0, col0, lane, acc);
#pragma unroll
  for (int t = 0; t < 4; ++t) {
    const int col = col0 + t * 16 + ln;
    const float bv = bias[col];
#pragma unroll
    for (int r = 0; r < 8; ++r)
      out[(size_t)(row0 + r + 8 * hi) * EE + col] = acc[t][r] + bv;
  }
}

// ---- kernel: Vt[bh][d][l] = V[b,l,h*64+d] (bf16) -----------------------------
__global__ __launch_bounds__(256) void k_transpose_v(const bf16_t* __restrict__ Vb,
                                                     bf16_t* __restrict__ Vt) {
  int idx = blockIdx.x * 256 + threadIdx.x;   // B*H*DH*L = 2M
  int l    = idx & (LL - 1);
  int rest = idx >> 10;
  int d    = rest & (DHH - 1);
  int bh   = rest >> 6;
  int b = bh >> 4, h = bh & (HH - 1);
  Vt[idx] = Vb[((size_t)(b * LL + l)) * EE + h * DHH + d];
}

// ---- kernel: QR[bh][l][r] = sum_d Q[b,h,l,d] * relk[r][d]  ([L,64]x[64,128]) -
__global__ __launch_bounds__(128) void k_qrel(const bf16_t* __restrict__ Qb,
                                              const float* __restrict__ relk,
                                              float* __restrict__ QR) {
  const int lane = threadIdx.x & 31, wave = threadIdx.x >> 5;
  const int hi = lane >> 4, ln = lane & 15;
  const int bh = blockIdx.z, b = bh >> 4, h = bh & 15;
  const int row0 = blockIdx.y * 64 + wave * 16;
  const int col0 = blockIdx.x * 64;
  fx8 acc[4] = {};
#pragma unroll
  for (int k0 = 0; k0 < DHH; k0 += 32) {
    bfx16 af = load_afrag_bf16(
        Qb + ((size_t)(b * LL + row0 + ln)) * EE + h * DHH + k0 + 16 * hi);
#pragma unroll
    for (int t = 0; t < 4; ++t) {
      const int rr = col0 + t * 16 + ln;   // relative index column (N)
      bfx16 bfr;
      if (rr < RK) {
        bfr = load_bfrag_f32(relk + (size_t)rr * DHH + k0 + 16 * hi);
      } else {
#pragma unroll
        for (int j = 0; j < 16; ++j) bfr[j] = f2bf(0.f);
      }
      acc[t] = wmma32(af, bfr, acc[t]);
    }
  }
#pragma unroll
  for (int t = 0; t < 4; ++t)
#pragma unroll
    for (int r = 0; r < 8; ++r)
      QR[((size_t)bh * LL + row0 + r + 8 * hi) * RKP + col0 + t * 16 + ln] =
          acc[t][r];
}

// ---- kernel: fused scores (+rel bias, scale) -> LDS -> softmax -> weights ----
// One block = one (bh, 64-row l-strip). Scores strip lives in 256 KB LDS
// (CDNA5: up to 320 KB per workgroup). Each wave owns 16 rows end-to-end, so
// no block barriers are needed; weights hit HBM exactly once, post-softmax.
__global__ __launch_bounds__(128) void k_scores_softmax(
    const bf16_t* __restrict__ Qb, const bf16_t* __restrict__ Kb,
    const float* __restrict__ QR, float* __restrict__ Wt) {
  __shared__ float S[64 * LL];                 // 256 KB
  const int lane = threadIdx.x & 31, wave = threadIdx.x >> 5;
  const int hi = lane >> 4, ln = lane & 15;
  const int bh = blockIdx.y, b = bh >> 4, h = bh & 15;
  const int l0 = blockIdx.x * 64;              // block's l-strip base
  const int lw = l0 + wave * 16;               // wave's 16-row base

  // hoisted Q fragments (depend only on k0)
  bfx16 aq[2];
#pragma unroll
  for (int kk = 0; kk < 2; ++kk)
    aq[kk] = load_afrag_bf16(
        Qb + ((size_t)(b * LL + lw + ln)) * EE + h * DHH + kk * 32 + 16 * hi);

  // phase 1: 16 x 1024 score strip via WMMA, epilogue adds rel bias + scale
  for (int mt = 0; mt < 16; ++mt) {
    const int m0 = mt * 64;
    fx8 acc[4] = {};
#pragma unroll
    for (int kk = 0; kk < 2; ++kk) {
#pragma unroll
      for (int t = 0; t < 4; ++t) {
        const bf16_t* pb = Kb + ((size_t)(b * LL + m0 + t * 16 + ln)) * EE +
                           h * DHH + kk * 32 + 16 * hi;
        acc[t] = wmma32(aq[kk], load_bfrag_bf16(pb), acc[t]);
      }
    }
#pragma unroll
    for (int t = 0; t < 4; ++t) {
      const int m = m0 + t * 16 + ln;
#pragma unroll
      for (int r = 0; r < 8; ++r) {
        const int lr = wave * 16 + r + 8 * hi;   // row inside block strip
        const int l  = l0 + lr;
        float s = acc[t][r];
        const int rr = RM - 1 + l - m;
        if (rr >= 0 && rr < RK) s += QR[((size_t)bh * LL + l) * RKP + rr];
        S[lr * LL + m] = s * 0.03125f;           // 1/sqrt(1024)
      }
    }
  }

  // phase 2: per-wave softmax on its own 16 rows (lane-strided, conflict-free)
  for (int r = 0; r < 16; ++r) {
    const int lr = wave * 16 + r;
    float* row = S + lr * LL;
    float mx = -3.402823466e38f;
    for (int k = 0; k < 32; ++k) mx = fmaxf(mx, row[k * 32 + lane]);
#pragma unroll
    for (int i = 16; i > 0; i >>= 1) mx = fmaxf(mx, __shfl_xor(mx, i, 32));
    float sum = 0.f;
    for (int k = 0; k < 32; ++k) {
      float e = __expf(row[k * 32 + lane] - mx);
      sum += e;
      row[k * 32 + lane] = e;
    }
#pragma unroll
    for (int i = 16; i > 0; i >>= 1) sum += __shfl_xor(sum, i, 32);
    const float inv = 1.0f / sum;
    float* out = Wt + ((size_t)bh * LL + l0 + lr) * LL;
    for (int k = 0; k < 32; ++k) out[k * 32 + lane] = row[k * 32 + lane] * inv;
  }
}

// ---- kernel: ctx = W·V + skew(W)·relv, stored merged-head [B,L,E] f32 --------
__global__ __launch_bounds__(128) void k_context(const float* __restrict__ Wt,
                                                 const bf16_t* __restrict__ Vt,
                                                 const float* __restrict__ relv,
                                                 float* __restrict__ ctx) {
  const int lane = threadIdx.x & 31, wave = threadIdx.x >> 5;
  const int hi = lane >> 4, ln = lane & 15;
  const int bh = blockIdx.y, b = bh >> 4, h = bh & 15;
  const int l0 = blockIdx.x * 64 + wave * 16;
  fx8 acc[4] = {};
  const float* wrow = Wt + ((size_t)bh * LL + l0 + ln) * LL;  // this lane's l-row
  // main contraction over m = 0..1023
  for (int m0 = 0; m0 < LL; m0 += 32) {
    bfx16 af = load_afrag_f32(wrow + m0 + 16 * hi);
#pragma unroll
    for (int t = 0; t < 4; ++t) {
      const bf16_t* pb = Vt + ((size_t)bh * DHH + t * 16 + ln) * LL + m0 + 16 * hi;
      acc[t] = wmma32(af, load_bfrag_bf16(pb), acc[t]);
    }
  }
  // relative-value term: out[l,d] += sum_r w[l, l+63-r] * relv[r][d]
  const int l = l0 + ln;
#pragma unroll
  for (int r0 = 0; r0 < RKP; r0 += 32) {
    bfx16 af;
#pragma unroll
    for (int j = 0; j < 16; ++j) {
      const int r = r0 + j + 16 * hi;
      const int m = l + RM - 1 - r;
      float wv = 0.f;
      if (r < RK && m >= 0 && m < LL) wv = wrow[m];
      af[j] = f2bf(wv);
    }
#pragma unroll
    for (int t = 0; t < 4; ++t) {
      bfx16 bfr;
#pragma unroll
      for (int j = 0; j < 16; ++j) {
        const int r = r0 + j + 16 * hi;
        bfr[j] = (r < RK) ? f2bf(relv[(size_t)r * DHH + t * 16 + ln]) : f2bf(0.f);
      }
      acc[t] = wmma32(af, bfr, acc[t]);
    }
  }
#pragma unroll
  for (int t = 0; t < 4; ++t)
#pragma unroll
    for (int r = 0; r < 8; ++r)
      ctx[((size_t)(b * LL + l0 + r + 8 * hi)) * EE + h * DHH + t * 16 + ln] =
          acc[t][r];
}

extern "C" void kernel_launch(void* const* d_in, const int* in_sizes, int n_in,
                              void* d_out, int out_size, void* d_ws, size_t ws_size,
                              hipStream_t stream) {
  const float* queries = (const float*)d_in[0];
  const float* keys    = (const float*)d_in[1];
  const float* values  = (const float*)d_in[2];
  const float* Wq = (const float*)d_in[3];
  const float* bq = (const float*)d_in[4];
  const float* Wk = (const float*)d_in[5];
  const float* bk = (const float*)d_in[6];
  const float* Wv = (const float*)d_in[7];
  const float* bv = (const float*)d_in[8];
  const float* Wo = (const float*)d_in[9];
  const float* bo = (const float*)d_in[10];
  const float* relk = (const float*)d_in[11];
  const float* relv = (const float*)d_in[12];

  char* ws = (char*)d_ws;
  const size_t szWp  = (size_t)EE * EE * sizeof(bf16_t);          // 2 MB each
  const size_t szPrj = (size_t)BB * LL * EE * sizeof(bf16_t);     // 4 MB each
  bf16_t* WpQ = (bf16_t*)(ws);
  bf16_t* WpK = (bf16_t*)(ws + szWp);
  bf16_t* WpV = (bf16_t*)(ws + 2 * szWp);
  bf16_t* WpO = (bf16_t*)(ws + 3 * szWp);
  bf16_t* Qb  = (bf16_t*)(ws + 4 * szWp);
  bf16_t* Kb  = (bf16_t*)(ws + 4 * szWp + szPrj);
  bf16_t* Vb  = (bf16_t*)(ws + 4 * szWp + 2 * szPrj);
  bf16_t* Vtb = (bf16_t*)(ws + 4 * szWp + 3 * szPrj);
  float*  QR  = (float*)(ws + 4 * szWp + 4 * szPrj);              // 16 MB
  float*  ctx = (float*)(ws + 4 * szWp + 4 * szPrj +
                         (size_t)BB * HH * LL * RKP * sizeof(float)); // 8 MB

  float* outF = (float*)d_out;
  float* Wt   = outF + (size_t)BB * LL * EE;   // softmax weights output region

  // 1) pack the four ExE weight matrices into WMMA B-fragment order
  k_pack_w<<<256, 256, 0, stream>>>(Wq, WpQ);
  k_pack_w<<<256, 256, 0, stream>>>(Wk, WpK);
  k_pack_w<<<256, 256, 0, stream>>>(Wv, WpV);
  k_pack_w<<<256, 256, 0, stream>>>(Wo, WpO);

  // 2) Q/K/V projections (bf16 outputs)
  k_gemm_bf16<<<dim3(16, 32), 128, 0, stream>>>(queries, WpQ, bq, Qb);
  k_gemm_bf16<<<dim3(16, 32), 128, 0, stream>>>(keys,    WpK, bk, Kb);
  k_gemm_bf16<<<dim3(16, 32), 128, 0, stream>>>(values,  WpV, bv, Vb);
  k_transpose_v<<<8192, 256, 0, stream>>>(Vb, Vtb);

  // 3) QR = Q · relk^T  (per head, [L,64]x[64,128])
  k_qrel<<<dim3(2, 16, 32), 128, 0, stream>>>(Qb, relk, QR);

  // 4+5) fused scores + rel bias + scale + softmax -> weights (single HBM write)
  k_scores_softmax<<<dim3(16, 32), 128, 0, stream>>>(Qb, Kb, QR, Wt);

  // 6) context = W·V + skew(W)·relv
  k_context<<<dim3(16, 32), 128, 0, stream>>>(Wt, Vtb, relv, ctx);

  // 7) output projection -> d_out[0 .. B*L*E)
  k_gemm_f32<<<dim3(16, 32), 128, 0, stream>>>(ctx, WpO, bo, outF);
}
